// ormamba_87136296501515
// MI455X (gfx1250) — matrix operand order repaired and checked by reading, not compiled, not run on hardware
//
#include <hip/hip_runtime.h>
#include <hip/hip_bf16.h>
#include <math.h>

// ---------------- Mamba dims (compile-time constants from reference) --------
#define BB       4
#define LL       2048
#define D_MODEL  256
#define D_STATE  16
#define D_CONV   4
#define D_INNER  512
#define DT_RANK  16
#define ML       (BB * LL)            // 8192 rows for all GEMMs
#define XDBL_LD  64                   // padded row stride of x_dbl (dt|B|C|pad)

typedef __attribute__((ext_vector_type(16))) __bf16 v16bf;
typedef __attribute__((ext_vector_type(8)))  float  v8f;

__device__ __forceinline__ float softplusf(float x) {
    return (x > 20.0f) ? x : log1pf(__expf(x));
}
__device__ __forceinline__ float siluf(float x) {
    return x / (1.0f + __expf(-x));
}

__device__ __forceinline__ v16bf pack_bf16(float4 a, float4 b, float4 c, float4 d) {
    v16bf r;
    r[0]  = (__bf16)a.x; r[1]  = (__bf16)a.y; r[2]  = (__bf16)a.z; r[3]  = (__bf16)a.w;
    r[4]  = (__bf16)b.x; r[5]  = (__bf16)b.y; r[6]  = (__bf16)b.z; r[7]  = (__bf16)b.w;
    r[8]  = (__bf16)c.x; r[9]  = (__bf16)c.y; r[10] = (__bf16)c.z; r[11] = (__bf16)c.w;
    r[12] = (__bf16)d.x; r[13] = (__bf16)d.y; r[14] = (__bf16)d.z; r[15] = (__bf16)d.w;
    return r;
}

union Bfrag { uint4 u[2]; v16bf v; };

// A fragment 16x32 bf16 (ISA 7.12.2): lane m = lane%16;
// k = k0 + 8*(lane>=16) + {0..7, 16..23} -> float4s at kb/4 + {0,1,4,5}
__device__ __forceinline__ v16bf load_afrag(const float4* __restrict__ Arow, int ai) {
    return pack_bf16(Arow[ai], Arow[ai + 1], Arow[ai + 4], Arow[ai + 5]);
}

// ---------------------------------------------------------------------------
// WMMA GEMM:  C[m][n] = sum_k A[m][k] * Wb[n][k]  (+ bias / softplus)
//   A  : row-major [M x lda] fp32
//   Wb : row-major [N x K]   bf16 (pre-converted, tile-exact)
// Dims tile-exact: M % 128 == 0, N % 64 == 0, K % 32 == 0.
// Block = 256 threads = 8 waves; block tile 128(M) x 64(N);
// wave tile 16(M) x 64(N), 4 accumulators; K stepped 32.
// Software-pipelined: iteration k+1 fragment loads issue before the 4 WMMAs
// of iteration k, keeping ~12 b128 loads in flight across each WMMA burst.
// mode 0: plain   mode 1: softplus(x + bias[n])
// ---------------------------------------------------------------------------
__global__ __launch_bounds__(256)
void wmma_gemm_bf16(const float* __restrict__ A, const __bf16* __restrict__ Wb,
                    float* __restrict__ C, const float* __restrict__ bias,
                    int N, int K, int lda, int mode)
{
    const int lane  = threadIdx.x & 31;
    const int wave  = threadIdx.x >> 5;
    const int row0  = blockIdx.x * 128 + wave * 16;
    const int cb    = blockIdx.y * 64;

    const int nLane  = lane & 15;
    const int aShift = (lane >> 4) << 3;   // 0 / 8  (A frag k-interleave)
    const int bShift = (lane >> 4) << 3;   // 0 / 8  (B frag k-interleave)
    const int mA     = row0 + nLane;

    const float4* Arow = (const float4*)(A + (size_t)mA * lda);
    const uint4*  Wrow[4];
    #pragma unroll
    for (int t = 0; t < 4; ++t)
        Wrow[t] = (const uint4*)(Wb + (size_t)(cb + t * 16 + nLane) * K);

    v8f acc[4] = {};

    // ---- prologue: load fragments for k0 = 0 ----
    v16bf af = load_afrag(Arow, aShift >> 2);
    Bfrag bf[4];
    #pragma unroll
    for (int t = 0; t < 4; ++t) {
        bf[t].u[0] = Wrow[t][bShift >> 3];
        bf[t].u[1] = Wrow[t][(bShift >> 3) + 2];
    }

    // ---- pipelined main loop: load(k0) then compute(k0-32) ----
    #pragma unroll 2
    for (int k0 = 32; k0 < K; k0 += 32) {
        __builtin_prefetch(A + (size_t)mA * lda + k0 + 128, 0, 0); // speculative

        v16bf afn = load_afrag(Arow, (k0 + aShift) >> 2);
        Bfrag bfn[4];
        const int bi = (k0 + bShift) >> 3;
        #pragma unroll
        for (int t = 0; t < 4; ++t) {
            bfn[t].u[0] = Wrow[t][bi];
            bfn[t].u[1] = Wrow[t][bi + 2];
        }

        #pragma unroll
        for (int t = 0; t < 4; ++t)
            acc[t] = __builtin_amdgcn_wmma_f32_16x16x32_bf16(
                         false, af, false, bf[t].v, (short)0, acc[t], false, false);

        af = afn;
        #pragma unroll
        for (int t = 0; t < 4; ++t) bf[t] = bfn[t];
    }

    // ---- epilogue compute (last K-step) ----
    #pragma unroll
    for (int t = 0; t < 4; ++t)
        acc[t] = __builtin_amdgcn_wmma_f32_16x16x32_bf16(
                     false, af, false, bf[t].v, (short)0, acc[t], false, false);

    // store D (f32 C/D layout: n = lane%16; m = v + 8*(lane>=16))
    const int mOff = (lane >> 4) << 3;
    #pragma unroll
    for (int t = 0; t < 4; ++t) {
        const int n = cb + t * 16 + nLane;
        const float bv = (mode == 1) ? bias[n] : 0.0f;
        #pragma unroll
        for (int v = 0; v < 8; ++v) {
            const int m = row0 + mOff + v;
            float val = acc[t][v];
            if (mode == 1) val = softplusf(val + bv);
            C[(size_t)m * N + n] = val;
        }
    }
}

// ---------------------------------------------------------------------------
// One-shot weight prep: convert all Linear weights to bf16 (zero-padding the
// two awkward ones to tile-exact shapes):
//   W_in   [1024,256] -> Wib  [1024,256] bf16
//   W_xproj  [48,512] -> Wxb    [64,512] bf16 (rows 48..63 = 0)
//   W_dt    [512, 16] -> Wdtb  [512, 32] bf16 (cols 16..31 = 0)
//   W_out  [256, 512] -> Wob  [256, 512] bf16
// ---------------------------------------------------------------------------
__global__ void prep_weights_kernel(const float* __restrict__ Win,
                                    const float* __restrict__ Wx,
                                    const float* __restrict__ Wdt,
                                    const float* __restrict__ Wout,
                                    __bf16* __restrict__ Wib,
                                    __bf16* __restrict__ Wxb,
                                    __bf16* __restrict__ Wdtb,
                                    __bf16* __restrict__ Wob)
{
    int i = blockIdx.x * blockDim.x + threadIdx.x;
    if (i < 1024 * 256) { Wib[i] = (__bf16)Win[i]; return; }
    i -= 1024 * 256;
    if (i < 64 * 512) {
        int n = i >> 9, k = i & 511;
        Wxb[i] = (__bf16)((n < 48) ? Wx[n * 512 + k] : 0.0f);
        return;
    }
    i -= 64 * 512;
    if (i < 512 * 32) {
        int n = i >> 5, k = i & 31;
        Wdtb[i] = (__bf16)((k < 16) ? Wdt[n * 16 + k] : 0.0f);
        return;
    }
    i -= 512 * 32;
    if (i < 256 * 512) Wob[i] = (__bf16)Wout[i];
}

// ---------------------------------------------------------------------------
// Causal depthwise conv1d (k=4) + bias + SiLU over u-half of xz.
// ---------------------------------------------------------------------------
__global__ void conv_silu_kernel(const float* __restrict__ xz,
                                 const float* __restrict__ conv_w,
                                 const float* __restrict__ conv_b,
                                 float* __restrict__ u, int total)
{
    int idx = blockIdx.x * blockDim.x + threadIdx.x;
    if (idx >= total) return;
    int d  = idx & (D_INNER - 1);
    int bl = idx >> 9;                // b*L + l
    int l  = bl & (LL - 1);
    float acc = conv_b[d];
    #pragma unroll
    for (int j = 0; j < D_CONV; ++j) {
        int ls = l - (D_CONV - 1) + j;
        if (ls >= 0)
            acc += xz[(size_t)(bl - (D_CONV - 1 - j)) * (2 * D_INNER) + d]
                   * conv_w[d * D_CONV + j];
    }
    u[idx] = siluf(acc);
}

// ---------------------------------------------------------------------------
// Selective scan, lane-per-(channel,state); y reduced by 4 xor-shuffles
// inside each 16-lane state group (wave32-safe). Fused skip + gate epilogue.
// ---------------------------------------------------------------------------
__global__ __launch_bounds__(256)
void scan_kernel(const float* __restrict__ delta,
                 const float* __restrict__ u,
                 const float* __restrict__ xdbl,   // [B*L, 64]: dt|B|C|pad
                 const float* __restrict__ xz,     // z at cols [512,1024)
                 const float* __restrict__ A_log,
                 const float* __restrict__ Dskip,
                 float* __restrict__ yg)
{
    const int tid   = threadIdx.x;
    const int n     = tid & 15;
    const int dloc  = tid >> 4;                       // 0..15
    const int b     = blockIdx.x >> 5;                // 32 blocks per batch
    const int dtile = blockIdx.x & 31;
    const int d     = dtile * 16 + dloc;

    const float a  = -__expf(A_log[d * D_STATE + n]);
    const float Dd = Dskip[d];
    float h = 0.0f;

    const size_t rowBase = (size_t)b * LL;
    for (int l = 0; l < LL; ++l) {
        const size_t bl  = rowBase + l;
        const size_t off = bl * D_INNER + d;
        float dv = delta[off];                        // broadcast in 16-lane grp
        float uv = u[off];
        float Bn = xdbl[bl * XDBL_LD + DT_RANK + n];
        float Cn = xdbl[bl * XDBL_LD + DT_RANK + D_STATE + n];
        h = __expf(dv * a) * h + (dv * uv) * Bn;
        float p = h * Cn;
        p += __shfl_xor(p, 1, 32);
        p += __shfl_xor(p, 2, 32);
        p += __shfl_xor(p, 4, 32);
        p += __shfl_xor(p, 8, 32);
        if (n == 0) {
            float z = xz[bl * (2 * D_INNER) + D_INNER + d];
            yg[off] = (p + uv * Dd) * siluf(z);
        }
    }
}

// ---------------------------------------------------------------------------
extern "C" void kernel_launch(void* const* d_in, const int* in_sizes, int n_in,
                              void* d_out, int out_size, void* d_ws, size_t ws_size,
                              hipStream_t stream)
{
    const float* x      = (const float*)d_in[0];  // [B,L,256]
    const float* W_in   = (const float*)d_in[1];  // [1024,256]
    const float* conv_w = (const float*)d_in[2];  // [512,1,4]
    const float* conv_b = (const float*)d_in[3];  // [512]
    const float* W_xprj = (const float*)d_in[4];  // [48,512]
    const float* W_dt   = (const float*)d_in[5];  // [512,16]
    const float* b_dt   = (const float*)d_in[6];  // [512]
    const float* A_log  = (const float*)d_in[7];  // [512,16]
    const float* Dskip  = (const float*)d_in[8];  // [512]
    const float* W_out  = (const float*)d_in[9];  // [256,512]
    float* out = (float*)d_out;                   // [B,L,256]

    char* ws = (char*)d_ws;
    float* xz    = (float*)ws;  ws += (size_t)ML * 2 * D_INNER * sizeof(float); // 32 MB
    float* u     = (float*)ws;  ws += (size_t)ML * D_INNER * sizeof(float);     // 16 MB
    float* xdbl  = (float*)ws;  ws += (size_t)ML * XDBL_LD * sizeof(float);     //  2 MB
    float* delta = (float*)ws;  ws += (size_t)ML * D_INNER * sizeof(float);     // 16 MB
    float* yg    = (float*)ws;  ws += (size_t)ML * D_INNER * sizeof(float);     // 16 MB
    __bf16* Wib  = (__bf16*)ws; ws += (size_t)1024 * 256 * sizeof(__bf16);      // 512 KB
    __bf16* Wxb  = (__bf16*)ws; ws += (size_t)64 * 512 * sizeof(__bf16);        //  64 KB
    __bf16* Wdtb = (__bf16*)ws; ws += (size_t)512 * 32 * sizeof(__bf16);        //  32 KB
    __bf16* Wob  = (__bf16*)ws;                                                 // 256 KB

    const dim3 blk(256);

    // 0) convert + pad all weights to bf16 (one shot)
    {
        int total = 1024 * 256 + 64 * 512 + 512 * 32 + 256 * 512;
        prep_weights_kernel<<<(total + 255) / 256, blk, 0, stream>>>(
            W_in, W_xprj, W_dt, W_out, Wib, Wxb, Wdtb, Wob);
    }

    // 1) in_proj: xz = x @ W_in^T      M=8192 N=1024 K=256
    wmma_gemm_bf16<<<dim3(ML / 128, 1024 / 64), blk, 0, stream>>>(
        x, Wib, xz, nullptr, 2 * D_INNER, D_MODEL, D_MODEL, 0);

    // 2) depthwise causal conv + SiLU -> u
    {
        int total = ML * D_INNER;
        conv_silu_kernel<<<(total + 255) / 256, blk, 0, stream>>>(
            xz, conv_w, conv_b, u, total);
    }

    // 3) x_proj: xdbl = u @ Wxb^T     M=8192 N=64(padded) K=512
    wmma_gemm_bf16<<<dim3(ML / 128, 1), blk, 0, stream>>>(
        u, Wxb, xdbl, nullptr, XDBL_LD, D_INNER, D_INNER, 0);

    // 4) dt_proj + softplus: delta = softplus(xdbl[:, :16] @ Wdtb^T + b_dt)
    //    M=8192 N=512 K=32(padded; zero weights kill cols 16..31), lda=64
    wmma_gemm_bf16<<<dim3(ML / 128, D_INNER / 64), blk, 0, stream>>>(
        xdbl, Wdtb, delta, b_dt, D_INNER, 32, XDBL_LD, 1);

    // 5) selective scan + skip + gate -> yg
    scan_kernel<<<dim3(BB * (D_INNER / 16)), blk, 0, stream>>>(
        delta, u, xdbl, xz, A_log, Dskip, yg);

    // 6) out_proj: out = yg @ W_out^T  M=8192 N=256 K=512
    wmma_gemm_bf16<<<dim3(ML / 128, D_MODEL / 64), blk, 0, stream>>>(
        yg, Wob, out, nullptr, D_MODEL, D_INNER, D_INNER, 0);
}
